// MultiLayerReranker_35639638622828
// MI455X (gfx1250) — matrix operand order repaired and checked by reading, not compiled, not run on hardware
//
#include <hip/hip_runtime.h>
#include <hip/hip_bf16.h>

// ---------------------------------------------------------------------------
// DTW similarity, MI455X (gfx1250, wave32, WMMA).
//   seq1: (64, 128, 1024) f32   seq2: (64, 256, 1024) f32   out: (64,) f32
// Pipeline (2 kernels, norms fused into the GEMM):
//   1) gemm_kernel : sim[b] = seq1[b] @ seq2[b]^T via v_wmma_f32_16x16x32_bf16
//                    (fp32 -> bf16 during LDS staging); row sums-of-squares
//                    accumulated during staging, reduced with ds_add_f32 LDS
//                    atomics -> reciprocal norms written as a side output.
//   2) dtw_kernel  : wave-synchronous anti-diagonal DP. One wave32 per batch,
//                    4 rows per lane; the cross-lane dependency moves through
//                    __shfl_up (no barriers at all); sim values prefetched one
//                    diagonal ahead; dist = 1 - sim*rn1*rn2 fused.
// Workspace layout (floats): sim[64*128*256] | rn1[64*128] | rn2[64*256]
// ---------------------------------------------------------------------------

typedef __attribute__((ext_vector_type(16))) __bf16 v16bf;
typedef __attribute__((ext_vector_type(8)))  __bf16 v8bf;
typedef __attribute__((ext_vector_type(8)))  float  v8f;

#define BATCH 64
#define MDIM 128
#define NDIM 256
#define KDIM 1024
#define BLK_M 128               // full M per block
#define BLK_N 128               // two N-tiles per batch
#define BLK_K 64                // two WMMA K-steps per stage
#define LDS_PAD 8
#define LDROW (BLK_K + LDS_PAD) // 72 bf16 = 144B row stride (16B aligned)

// ----------------------------------------------------------------- gemm ----
// Block: 512 threads (16 waves as 4x4). C-tile 128x128; wave-tile 32x32
// (2x2 fragments of 16x16). K staged 64 at a time through LDS as bf16.
// Row norms of the staged fp32 data are computed on the fly.
__global__ __launch_bounds__(512) void dtw_gemm_kernel(
    const float* __restrict__ seq1, const float* __restrict__ seq2,
    float* __restrict__ sim, float* __restrict__ rn1,
    float* __restrict__ rn2) {
  __shared__ __bf16 lA[BLK_M][LDROW];
  __shared__ __bf16 lB[BLK_N][LDROW];
  __shared__ float  sqA[BLK_M];
  __shared__ float  sqB[BLK_N];

  const int b   = blockIdx.z;
  const int nt  = blockIdx.x;       // 0..1
  const int tid = threadIdx.x;
  const int wave = tid >> 5;        // 0..15
  const int lane = tid & 31;
  const int lr   = lane & 15;       // 0..15
  const int lh   = lane >> 4;       // half-wave select
  const int wm   = wave >> 2;       // 0..3  (M)
  const int wn   = wave & 3;        // 0..3  (N)

  if (tid < BLK_M) sqA[tid] = 0.0f;
  if (tid < BLK_N) sqB[tid] = 0.0f;

  const float* Ab = seq1 + (size_t)b * MDIM * KDIM;
  const float* Bb = seq2 + ((size_t)b * NDIM + nt * BLK_N) * KDIM;

  // staging ownership: thread handles (row = tid/16 + 32r, kc = (tid%16)*4)
  const int srow = tid >> 4;            // 0..31
  const int skc  = (tid & 15) << 2;     // 0,4,...,60

  v8f   acc[2][2] = {};
  float pa[4] = {0.f, 0.f, 0.f, 0.f};   // per-thread row sumsq partials (A)
  float pb[4] = {0.f, 0.f, 0.f, 0.f};   // per-thread row sumsq partials (B)

  for (int k0 = 0; k0 < KDIM; k0 += BLK_K) {
    // --- stage fp32 -> bf16 tiles into LDS, accumulating sums-of-squares --
    #pragma unroll
    for (int r = 0; r < 4; ++r) {               // A: 128x64 = 2048 float4
      const int row = srow + 32 * r;
      float4 v = *(const float4*)(Ab + (size_t)row * KDIM + k0 + skc);
      pa[r] += v.x * v.x + v.y * v.y + v.z * v.z + v.w * v.w;
      __bf16* d = &lA[row][skc];
      d[0] = (__bf16)v.x; d[1] = (__bf16)v.y;
      d[2] = (__bf16)v.z; d[3] = (__bf16)v.w;
    }
    #pragma unroll
    for (int r = 0; r < 4; ++r) {               // B: 128x64 = 2048 float4
      const int row = srow + 32 * r;
      float4 v = *(const float4*)(Bb + (size_t)row * KDIM + k0 + skc);
      pb[r] += v.x * v.x + v.y * v.y + v.z * v.z + v.w * v.w;
      __bf16* d = &lB[row][skc];
      d[0] = (__bf16)v.x; d[1] = (__bf16)v.y;
      d[2] = (__bf16)v.z; d[3] = (__bf16)v.w;
    }
    if (k0 + BLK_K < KDIM) {                    // -> global_prefetch_b8
      __builtin_prefetch(Ab + (size_t)srow * KDIM + k0 + BLK_K + skc, 0, 1);
      __builtin_prefetch(Bb + (size_t)srow * KDIM + k0 + BLK_K + skc, 0, 1);
    }
    __syncthreads();

    // --- 2 WMMA K-steps per stage, fragments per CDNA5 16-bit layouts ----
    #pragma unroll
    for (int kt = 0; kt < 2; ++kt) {
      // A (16x32): lane half 0 -> K[0..7],[16..23]; half 1 -> K[8..15],[24..31]
      v16bf afrag[2], bfrag[2];
      #pragma unroll
      for (int im = 0; im < 2; ++im) {
        const int row = wm * 32 + im * 16 + lr;
        v8bf lo = *(const v8bf*)&lA[row][kt * 32 + lh * 8];
        v8bf hi = *(const v8bf*)&lA[row][kt * 32 + lh * 8 + 16];
        afrag[im] = __builtin_shufflevector(lo, hi,
            0, 1, 2, 3, 4, 5, 6, 7, 8, 9, 10, 11, 12, 13, 14, 15);
      }
      // B (32x16): lane half 0 -> K[0..7],[16..23]; half 1 -> K[8..15],[24..31], N = lr
      #pragma unroll
      for (int in = 0; in < 2; ++in) {
        const int row = wn * 32 + in * 16 + lr;
        v8bf lo = *(const v8bf*)&lB[row][kt * 32 + lh * 8];
        v8bf hi = *(const v8bf*)&lB[row][kt * 32 + lh * 8 + 16];
        bfrag[in] = __builtin_shufflevector(lo, hi,
            0, 1, 2, 3, 4, 5, 6, 7, 8, 9, 10, 11, 12, 13, 14, 15);
      }
      #pragma unroll
      for (int im = 0; im < 2; ++im)
        #pragma unroll
        for (int in = 0; in < 2; ++in)
          acc[im][in] = __builtin_amdgcn_wmma_f32_16x16x32_bf16(
              false, afrag[im], false, bfrag[in],
              (short)0, acc[im][in], false, false);
    }
    __syncthreads();
  }

  // --- reduce row sums-of-squares via LDS float atomics (ds_add_f32) -----
  #pragma unroll
  for (int r = 0; r < 4; ++r) {
    atomicAdd(&sqA[srow + 32 * r], pa[r]);
    atomicAdd(&sqB[srow + 32 * r], pb[r]);
  }
  __syncthreads();

  if (tid < BLK_M) {
    if (nt == 0) rn1[b * MDIM + tid] = rsqrtf(sqA[tid]);
    rn2[b * NDIM + nt * BLK_N + tid] = rsqrtf(sqB[tid]);
  }

  // --- write raw similarity (norm scaling fused into DTW kernel) ---------
  // C layout: VGPR r -> M = base + lh*8 + r, N = base + lr
  float* Cb = sim + (size_t)b * MDIM * NDIM + nt * BLK_N;
  #pragma unroll
  for (int im = 0; im < 2; ++im) {
    #pragma unroll
    for (int in = 0; in < 2; ++in) {
      const int mb = wm * 32 + im * 16 + lh * 8;
      const int nn = wn * 32 + in * 16 + lr;
      #pragma unroll
      for (int r = 0; r < 8; ++r)
        Cb[(size_t)(mb + r) * NDIM + nn] = acc[im][in][r];
    }
  }
}

// ------------------------------------------------------------------ dtw ----
// One wave32 per batch; lane L owns rows 4L..4L+3. Anti-diagonal wavefront:
// within a diagonal there are no dependencies, so each step needs only the
// previous two diagonals. The single cross-lane value (row 4L-1) moves via
// __shfl_up; no barriers anywhere. sim is prefetched one diagonal ahead.
__global__ __launch_bounds__(32) void dtw_dp_kernel(
    const float* __restrict__ sim, const float* __restrict__ rn1,
    const float* __restrict__ rn2, float* __restrict__ out) {
  __shared__ float R2s[NDIM];
  const float INF = 1e30f;
  const int b = blockIdx.x;
  const int L = threadIdx.x;          // lane 0..31

  const float* Sb = sim + (size_t)b * MDIM * NDIM;
  const float* Srow[4];
  float r1v[4], prev[4], prev2[4], nc[4];
  #pragma unroll
  for (int r = 0; r < 4; ++r) {
    const int row = 4 * L + r;
    Srow[r] = Sb + (size_t)row * NDIM;
    r1v[r]  = rn1[b * MDIM + row];
    prev[r] = INF;
    prev2[r] = INF;
    nc[r] = 0.0f;
  }
  // cache rn2 row in LDS (coalesced, once)
  #pragma unroll
  for (int t = 0; t < NDIM / 32; ++t)
    R2s[L + 32 * t] = rn2[b * NDIM + L + 32 * t];
  __syncthreads();   // single-wave WG: lowers to waitcnt fence (barrier->NOP)

  // preload dist source for diagonal 0 (only cell (0,0) exists)
  if (L == 0) nc[0] = Srow[0][0];

  float val = INF;
  for (int d = 0; d < MDIM + NDIM - 1; ++d) {
    // cross-lane: previous lane's row 3 values for this lane's row 0
    float up3 = __shfl_up(prev[3], 1, 32);
    float dg3 = __shfl_up(prev2[3], 1, 32);
    if (L == 0) { up3 = INF; dg3 = INF; }

    float cur[4];
    #pragma unroll
    for (int r = 0; r < 4; ++r) {
      const int row = 4 * L + r;
      const int j = d - row;
      float v = INF;
      if (j >= 0 && j < NDIM) {
        const float c  = 1.0f - nc[r] * r1v[r] * R2s[j];
        const float up = (r == 0) ? up3 : prev[r - 1];    // D[row-1][j]
        const float lf = prev[r];                          // D[row][j-1]
        const float dg = (r == 0) ? dg3 : prev2[r - 1];   // D[row-1][j-1]
        const float m  = fminf(fminf(up, lf), dg);
        v = (m > 1e29f) ? c : c + m;   // (0,0) has no predecessor
        val = v;
      }
      cur[r] = v;
    }

    // prefetch next diagonal's sim values (latency hidden behind this step)
    #pragma unroll
    for (int r = 0; r < 4; ++r) {
      const int jn = d + 1 - (4 * L + r);
      if (jn >= 0 && jn < NDIM) nc[r] = Srow[r][jn];
    }

    #pragma unroll
    for (int r = 0; r < 4; ++r) { prev2[r] = prev[r]; prev[r] = cur[r]; }
  }

  // final cell (127, 255) lives in lane 31, row 3, diagonal 382
  if (L == 31)
    out[b] = 1.0f / (1.0f + val * (1.0f / (float)(MDIM + NDIM)));
}

// --------------------------------------------------------------- launch ----
extern "C" void kernel_launch(void* const* d_in, const int* in_sizes, int n_in,
                              void* d_out, int out_size, void* d_ws, size_t ws_size,
                              hipStream_t stream) {
  const float* seq1 = (const float*)d_in[0];
  const float* seq2 = (const float*)d_in[1];
  float* out = (float*)d_out;

  float* sim = (float*)d_ws;                                   // 64*128*256
  float* rn1 = sim + (size_t)BATCH * MDIM * NDIM;              // 64*128
  float* rn2 = rn1 + BATCH * MDIM;                             // 64*256

  dim3 ggrid(NDIM / BLK_N, 1, BATCH);                          // (2,1,64)
  dtw_gemm_kernel<<<ggrid, 512, 0, stream>>>(seq1, seq2, sim, rn1, rn2);

  dtw_dp_kernel<<<dim3(BATCH), 32, 0, stream>>>(sim, rn1, rn2, out);
}
